// TitansLayer_36060545417749
// MI455X (gfx1250) — compile-verified, matrix operand order checked by reading, not run on hardware
//
#include <hip/hip_runtime.h>
#include <hip/hip_bf16.h>

// ---------------------------------------------------------------------------
// Titans layer for MI455X (gfx1250, wave32, WMMA + TDM async tensor loads).
// B=16, T=1024, D=512, FF=2048, M=B*T=16384.
// ---------------------------------------------------------------------------

#define BB 16
#define TT 1024
#define DD 512
#define FFD 2048
#define MM (BB * TT)
#define NCHUNK 4
#define RI 128   // state rows per workgroup chunk (RI * NCHUNK == DD)
#define USE_TDM 1

typedef unsigned short u16;
typedef __attribute__((ext_vector_type(16))) __bf16 v16bf;
typedef __attribute__((ext_vector_type(8)))  __bf16 v8bf;
typedef __attribute__((ext_vector_type(8)))  float  v8f;
typedef __attribute__((ext_vector_type(4)))  unsigned int u32x4;
typedef __attribute__((ext_vector_type(8)))  int i32x8;
typedef __attribute__((ext_vector_type(4)))  int i32x4;

__device__ __forceinline__ u16 f2bfu(float f) {
    unsigned u = __builtin_bit_cast(unsigned, f);
    u += 0x7FFFu + ((u >> 16) & 1u);           // round-to-nearest-even
    return (u16)(u >> 16);
}

__device__ __forceinline__ float block_sum_256(float v, float* red, int tid) {
    red[tid] = v;
    __syncthreads();
#pragma unroll
    for (int s = 128; s > 0; s >>= 1) {
        if (tid < s) red[tid] += red[tid + s];
        __syncthreads();
    }
    float t = red[0];
    __syncthreads();
    return t;
}

__device__ __forceinline__ v16bf ld16(const u16* p) {
    v8bf lo = *(const v8bf*)p;
    v8bf hi = *(const v8bf*)(p + 16);
    v16bf o;
#pragma unroll
    for (int i = 0; i < 8; ++i) { o[i] = lo[i]; o[i + 8] = hi[i]; }
    return o;
}

// ---------------------------------------------------------------------------
// TDM: async 1-D tile load (nelem fp32) global -> LDS. Issued once per wave
// that executes it (EXEC ignored); tracked with TENSORcnt, in-order per wave.
// ---------------------------------------------------------------------------
#if USE_TDM
__device__ __forceinline__ void tdm_load_1d(unsigned lds_byte, const void* gaddr,
                                            unsigned nelem) {
    unsigned long long ga = (unsigned long long)(uintptr_t)gaddr;
    u32x4 g0 = {};
    g0[0] = 1u;                                          // count=1 (valid), user mode
    g0[1] = lds_byte;                                    // LDS byte address
    g0[2] = (unsigned)ga;                                // global addr [31:0]
    g0[3] = (unsigned)((ga >> 32) & 0x1FFFFFFu) | (2u << 30);  // [56:32] | type=2
    i32x8 g1 = {};
    g1[0] = (int)(2u << 16);                             // data_size = 4 bytes
    g1[1] = (int)((nelem & 0xFFFFu) << 16);              // tensor_dim0 lo16
    g1[2] = (int)((nelem >> 16) | (1u << 16));           // tensor_dim0 hi16 | tensor_dim1=1
    g1[3] = (int)((nelem & 0xFFFFu) << 16);              // tile_dim0 = nelem
    g1[4] = 1;                                           // tile_dim1 = 1
    g1[5] = (int)nelem;                                  // tensor_dim0_stride
    i32x4 z4 = {};
#if defined(__clang_major__) && (__clang_major__ >= 23)
    i32x8 z8 = {};
    __builtin_amdgcn_tensor_load_to_lds(g0, g1, z4, z4, z8, 0);
#else
    __builtin_amdgcn_tensor_load_to_lds(g0, g1, z4, z4, 0);
#endif
}
__device__ __forceinline__ unsigned lds_byte_off(const void* p) {
    return (unsigned)(uintptr_t)p;   // generic LDS address carries offset in [31:0]
}
#endif

// ---------------------------------------------------------------------------
// Weight convert + transpose: fp32 [K,N] -> bf16 [N,K]
// ---------------------------------------------------------------------------
__global__ __launch_bounds__(256) void wconv_kernel(const float* __restrict__ W,
                                                    u16* __restrict__ WT,
                                                    int K, int N) {
    int idx = blockIdx.x * 256 + threadIdx.x;
    if (idx >= K * N) return;
    int k = idx / N, n = idx - k * N;
    WT[(size_t)n * K + k] = f2bfu(W[idx]);
}

// ---------------------------------------------------------------------------
// LayerNorm (D=512, one row per 256-thread block) -> bf16 row.
// Optionally fused alpha/eta = sigmoid(xn . Wa + ba), sigmoid(xn . We + be).
// ---------------------------------------------------------------------------
__global__ __launch_bounds__(256) void ln_kernel(const float* __restrict__ x,
                                                 const float* __restrict__ g,
                                                 const float* __restrict__ bta,
                                                 u16* __restrict__ xnb,
                                                 const float* __restrict__ Wa,
                                                 const float* __restrict__ ba,
                                                 const float* __restrict__ We,
                                                 const float* __restrict__ be,
                                                 float* __restrict__ alpha,
                                                 float* __restrict__ eta) {
    __shared__ float red[256];
    int row = blockIdx.x, tid = threadIdx.x;
    const float* xr = x + (size_t)row * DD;
    float x0 = xr[tid], x1 = xr[tid + 256];

    float mean = block_sum_256(x0 + x1, red, tid) * (1.0f / (float)DD);
    float d0 = x0 - mean, d1 = x1 - mean;
    float var = block_sum_256(d0 * d0 + d1 * d1, red, tid) * (1.0f / (float)DD);
    float rstd = rsqrtf(var + 1e-5f);

    float n0 = d0 * rstd * g[tid] + bta[tid];
    float n1 = d1 * rstd * g[tid + 256] + bta[tid + 256];
    xnb[(size_t)row * DD + tid]       = f2bfu(n0);
    xnb[(size_t)row * DD + tid + 256] = f2bfu(n1);

    if (Wa) {
        float da = block_sum_256(n0 * Wa[tid] + n1 * Wa[tid + 256], red, tid);
        if (tid == 0) alpha[row] = 1.0f / (1.0f + __expf(-(da + ba[0])));
        float de = block_sum_256(n0 * We[tid] + n1 * We[tid + 256], red, tid);
        if (tid == 0) eta[row] = 1.0f / (1.0f + __expf(-(de + be[0])));
    }
}

// ---------------------------------------------------------------------------
// WMMA bf16 GEMM: out[M,N] = A[M,K](bf16) * W^T[N,K](bf16) + bias
// Each wave computes a 32x64 register tile (2 M-tiles x 4 N-tiles = 8 accs):
// 8 v_wmma per K-step on 6 operand tiles -> ~4x the L2 reuse of 1-tile waves.
// 4 waves/WG cover 32x256 of output. EPI: 0 = fp32 store, 1 = fp32 + residual,
// 2 = ReLU + bf16 store.
// ---------------------------------------------------------------------------
template <int EPI>
__global__ __launch_bounds__(128) void gemm_wmma(const u16* __restrict__ A,
                                                 const u16* __restrict__ W,
                                                 const float* __restrict__ bias,
                                                 const float* __restrict__ resid,
                                                 float* __restrict__ outF,
                                                 u16* __restrict__ outB,
                                                 int M, int N, int K) {
    int lane = threadIdx.x & 31;
    int wave = threadIdx.x >> 5;
    int m0 = blockIdx.x * 32;
    int n0 = (blockIdx.y * 4 + wave) * 64;
    if (n0 >= N || m0 >= M) return;

    int r  = lane & 15;   // A row / W col within tile; also D column
    int hb = lane >> 4;   // K half-select per WMMA 16-bit A/B layout

    const u16* a0 = A + (size_t)(m0 + r) * K + hb * 8;
    const u16* a1 = a0 + (size_t)16 * K;
    const u16* b0 = W + (size_t)(n0 + r) * K + hb * 8;
    const u16* b1 = b0 + (size_t)16 * K;
    const u16* b2 = b0 + (size_t)32 * K;
    const u16* b3 = b0 + (size_t)48 * K;

    v8f acc[2][4];
#pragma unroll
    for (int mi = 0; mi < 2; ++mi)
#pragma unroll
        for (int ni = 0; ni < 4; ++ni) {
            float bv = bias ? bias[n0 + ni * 16 + r] : 0.0f;
#pragma unroll
            for (int i = 0; i < 8; ++i) acc[mi][ni][i] = bv;
        }

    for (int kt = 0; kt < K; kt += 32) {
        __builtin_prefetch(a0 + kt + 256, 0, 0);
        __builtin_prefetch(b0 + kt + 256, 0, 0);
        v16bf av[2], bw[4];
        av[0] = ld16(a0 + kt);
        av[1] = ld16(a1 + kt);
        bw[0] = ld16(b0 + kt);
        bw[1] = ld16(b1 + kt);
        bw[2] = ld16(b2 + kt);
        bw[3] = ld16(b3 + kt);
#pragma unroll
        for (int mi = 0; mi < 2; ++mi)
#pragma unroll
            for (int ni = 0; ni < 4; ++ni)
                acc[mi][ni] = __builtin_amdgcn_wmma_f32_16x16x32_bf16(
                    false, av[mi], false, bw[ni], (short)0, acc[mi][ni], false, false);
    }

    // D layout: element (Mrow = vgpr + 8*hb, Ncol = lane&15)
#pragma unroll
    for (int mi = 0; mi < 2; ++mi)
#pragma unroll
        for (int ni = 0; ni < 4; ++ni)
#pragma unroll
            for (int i = 0; i < 8; ++i) {
                int mr = m0 + mi * 16 + i + 8 * hb;
                int nc = n0 + ni * 16 + r;
                size_t idx = (size_t)mr * N + nc;
                float v = acc[mi][ni][i];
                if (EPI == 1) v += resid[idx];
                if (EPI == 2) v = v > 0.0f ? v : 0.0f;
                if (EPI == 2) outB[idx] = f2bfu(v);
                else          outF[idx] = v;
            }
}

// ---------------------------------------------------------------------------
// Row-wise L2 normalize in place (matches F.normalize eps=1e-12)
// ---------------------------------------------------------------------------
__global__ __launch_bounds__(256) void l2norm_kernel(float* __restrict__ v) {
    __shared__ float red[256];
    int row = blockIdx.x, tid = threadIdx.x;
    float* vr = v + (size_t)row * DD;
    float a = vr[tid], b = vr[tid + 256];
    float ss = block_sum_256(a * a + b * b, red, tid);
    float s = 1.0f / fmaxf(sqrtf(ss), 1e-12f);
    vr[tid] = a * s;
    vr[tid + 256] = b * s;
}

// ---------------------------------------------------------------------------
// Delta-rule scan. Row-partitioned state: WG (chunk,b) owns rows
// [chunk*RI, chunk*RI+RI) of S[b] resident in LDS (RI*DD*4 = 256 KB, enabled
// by CDNA5's 320 KB/WGP LDS). pred/err/update are fully local; read_j
// partials are written out and reduced later (read never feeds the
// recurrence, so no cross-WG sync per step is needed).
// Per-step k/q/v rows are double-buffered into LDS by the Tensor Data Mover:
// wave 0 issues 3 TDM descriptors for step t+1, then
// s_wait_tensorcnt(3) (TDM is in-order per wave) + barrier, then computes t.
// ---------------------------------------------------------------------------
__global__ __launch_bounds__(256) void scan_kernel(const float* __restrict__ q,
                                                   const float* __restrict__ k,
                                                   const float* __restrict__ v,
                                                   const float* __restrict__ alpha,
                                                   const float* __restrict__ eta,
                                                   float* __restrict__ partials) {
    extern __shared__ float S[];            // RI * DD fp32 state tile
    __shared__ float kS[2][DD];
    __shared__ float qS[2][RI], vS[2][RI];
    __shared__ float eS[RI];
    __shared__ float ae[2];

    int tid = threadIdx.x;
    int chunk = blockIdx.x, b = blockIdx.y;
    int i0 = chunk * RI;
    int lane = tid & 31, wv = tid >> 5;

    for (int i = tid; i < RI * DD; i += 256) S[i] = 0.0f;
    __syncthreads();

    const size_t rowBase = (size_t)b * TT;

#if USE_TDM
    if (wv == 0) {   // prologue: stage t=0 into buffer 0
        size_t rr0 = rowBase * DD;
        tdm_load_1d(lds_byte_off(&kS[0][0]), k + rr0, DD);
        tdm_load_1d(lds_byte_off(&qS[0][0]), q + rr0 + i0, RI);
        tdm_load_1d(lds_byte_off(&vS[0][0]), v + rr0 + i0, RI);
    }
#endif

    for (int t = 0; t < TT; ++t) {
        int cur = t & 1;
#if USE_TDM
        if (wv == 0) {
            if (t + 1 < TT) {
                size_t rn = (rowBase + t + 1) * DD;
                int nxt = (t + 1) & 1;
                tdm_load_1d(lds_byte_off(&kS[nxt][0]), k + rn, DD);
                tdm_load_1d(lds_byte_off(&qS[nxt][0]), q + rn + i0, RI);
                tdm_load_1d(lds_byte_off(&vS[nxt][0]), v + rn + i0, RI);
                __builtin_amdgcn_s_wait_tensorcnt(3);  // step-t transfers done
            } else {
                __builtin_amdgcn_s_wait_tensorcnt(0);
            }
        }
#else
        {
            size_t rr = (rowBase + t) * DD;
            kS[cur][tid]       = k[rr + tid];
            kS[cur][tid + 256] = k[rr + tid + 256];
            if (tid < RI) {
                qS[cur][tid] = q[rr + i0 + tid];
                vS[cur][tid] = v[rr + i0 + tid];
            }
        }
#endif
        if (tid == 0) {
            ae[0] = alpha[rowBase + t];
            ae[1] = eta[rowBase + t];
        }
        __syncthreads();

        float a_ = ae[0], e_ = ae[1];

        // pred_i = sum_j S[i,j]*k_j ; err_i = eta*(v_i - pred_i)
        // wave wv handles rows wv*16 .. wv*16+15; full-wave shuffle reduce.
        for (int rr2 = 0; rr2 < 16; ++rr2) {
            int i = wv * 16 + rr2;
            const float* Srow = &S[i * DD];
            float p = 0.0f;
#pragma unroll
            for (int m = 0; m < 16; ++m)
                p += Srow[lane + 32 * m] * kS[cur][lane + 32 * m];
#pragma unroll
            for (int off = 16; off > 0; off >>= 1)
                p += __shfl_xor(p, off, 32);
            if (lane == 0) eS[i] = e_ * (vS[cur][i] - p);
        }
        __syncthreads();

        // read partial (from OLD state) + rank-1 decayed update.
        // Thread owns columns {tid, tid+256}; column-major walk over rows.
        float oma = 1.0f - a_;
        int j0 = tid, j1 = tid + 256;
        float k0 = kS[cur][j0], k1 = kS[cur][j1];
        float r0 = 0.0f, r1 = 0.0f;
        for (int i = 0; i < RI; ++i) {
            float ee = eS[i];
            float qi = qS[cur][i];
            float* Sr = &S[i * DD];
            float s0 = Sr[j0], s1 = Sr[j1];
            r0 += qi * s0;
            r1 += qi * s1;
            Sr[j0] = oma * s0 + ee * k0;
            Sr[j1] = oma * s1 + ee * k1;
        }
        float* pOut = partials + ((size_t)chunk * MM + rowBase + t) * DD;
        pOut[j0] = r0;
        pOut[j1] = r1;
        __syncthreads();   // compute-t done before wave0 issues t+2 DMAs
    }
}

// ---------------------------------------------------------------------------
// Sum NCHUNK read-partials -> bf16 rows for the Wo WMMA GEMM.
// ---------------------------------------------------------------------------
__global__ __launch_bounds__(256) void reduce_reads_kernel(const float* __restrict__ part,
                                                           u16* __restrict__ out) {
    size_t idx = (size_t)blockIdx.x * 256 + threadIdx.x;
    const size_t stride = (size_t)MM * DD;
    if (idx >= stride) return;
    float s = part[idx] + part[idx + stride] + part[idx + 2 * stride] + part[idx + 3 * stride];
    out[idx] = f2bfu(s);
}

// ---------------------------------------------------------------------------
// Host-side orchestration
// ---------------------------------------------------------------------------
extern "C" void kernel_launch(void* const* d_in, const int* in_sizes, int n_in,
                              void* d_out, int out_size, void* d_ws, size_t ws_size,
                              hipStream_t stream) {
    const float* x   = (const float*)d_in[0];
    const float* Wq  = (const float*)d_in[1];
    const float* Wk  = (const float*)d_in[2];
    const float* Wv  = (const float*)d_in[3];
    const float* Wa  = (const float*)d_in[4];
    const float* ba  = (const float*)d_in[5];
    const float* We  = (const float*)d_in[6];
    const float* be  = (const float*)d_in[7];
    const float* Wo  = (const float*)d_in[8];
    const float* bo  = (const float*)d_in[9];
    const float* W1  = (const float*)d_in[10];
    const float* b1  = (const float*)d_in[11];
    const float* W2  = (const float*)d_in[12];
    const float* b2  = (const float*)d_in[13];
    const float* g1  = (const float*)d_in[14];
    const float* bt1 = (const float*)d_in[15];
    const float* g2  = (const float*)d_in[16];
    const float* bt2 = (const float*)d_in[17];
    float* out = (float*)d_out;

    // bump allocator over d_ws
    char* wp = (char*)d_ws;
    auto alloc = [&](size_t bytes) -> void* {
        void* p = (void*)wp;
        wp += (bytes + 255) & ~(size_t)255;
        return p;
    };
    u16*   WqT   = (u16*)alloc((size_t)DD * DD * 2);
    u16*   WkT   = (u16*)alloc((size_t)DD * DD * 2);
    u16*   WvT   = (u16*)alloc((size_t)DD * DD * 2);
    u16*   WoT   = (u16*)alloc((size_t)DD * DD * 2);
    u16*   W1T   = (u16*)alloc((size_t)DD * FFD * 2);
    u16*   W2T   = (u16*)alloc((size_t)FFD * DD * 2);
    u16*   xnb   = (u16*)alloc((size_t)MM * DD * 2);
    float* alp   = (float*)alloc((size_t)MM * 4);
    float* eta   = (float*)alloc((size_t)MM * 4);
    float* qb    = (float*)alloc((size_t)MM * DD * 4);
    float* kb    = (float*)alloc((size_t)MM * DD * 4);
    float* vb    = (float*)alloc((size_t)MM * DD * 4);
    float* parts = (float*)alloc((size_t)NCHUNK * MM * DD * 4);
    u16*   rdb   = (u16*)alloc((size_t)MM * DD * 2);
    float* x1    = (float*)alloc((size_t)MM * DD * 4);
    u16*   x2b   = (u16*)alloc((size_t)MM * DD * 2);
    u16*   hb    = (u16*)alloc((size_t)MM * FFD * 2);
    (void)ws_size; (void)in_sizes; (void)n_in; (void)out_size;

    // 1) weight convert + transpose to bf16 [N,K]
    wconv_kernel<<<(DD * DD + 255) / 256, 256, 0, stream>>>(Wq, WqT, DD, DD);
    wconv_kernel<<<(DD * DD + 255) / 256, 256, 0, stream>>>(Wk, WkT, DD, DD);
    wconv_kernel<<<(DD * DD + 255) / 256, 256, 0, stream>>>(Wv, WvT, DD, DD);
    wconv_kernel<<<(DD * DD + 255) / 256, 256, 0, stream>>>(Wo, WoT, DD, DD);
    wconv_kernel<<<(DD * FFD + 255) / 256, 256, 0, stream>>>(W1, W1T, DD, FFD);
    wconv_kernel<<<(FFD * DD + 255) / 256, 256, 0, stream>>>(W2, W2T, FFD, DD);

    // 2) LN1 + fused alpha/eta
    ln_kernel<<<MM, 256, 0, stream>>>(x, g1, bt1, xnb, Wa, ba, We, be, alp, eta);

    // 3) q,k,v projections (WMMA bf16 -> fp32)
    dim3 gQ(MM / 32, DD / 256);
    gemm_wmma<0><<<gQ, 128, 0, stream>>>(xnb, WqT, nullptr, nullptr, qb, nullptr, MM, DD, DD);
    gemm_wmma<0><<<gQ, 128, 0, stream>>>(xnb, WkT, nullptr, nullptr, kb, nullptr, MM, DD, DD);
    gemm_wmma<0><<<gQ, 128, 0, stream>>>(xnb, WvT, nullptr, nullptr, vb, nullptr, MM, DD, DD);

    // 4) l2norm q,k
    l2norm_kernel<<<MM, 256, 0, stream>>>(qb);
    l2norm_kernel<<<MM, 256, 0, stream>>>(kb);

    // 5) delta-rule scan, 256 KB state tile in LDS per WG, TDM double-buffered
    scan_kernel<<<dim3(NCHUNK, BB), 256, RI * DD * sizeof(float), stream>>>(
        qb, kb, vb, alp, eta, parts);

    // 6) reduce read partials -> bf16
    reduce_reads_kernel<<<((size_t)MM * DD + 255) / 256, 256, 0, stream>>>(parts, rdb);

    // 7) attn projection + residual: x1 = x + reads @ Wo + bo
    gemm_wmma<1><<<gQ, 128, 0, stream>>>(rdb, WoT, bo, x, x1, nullptr, MM, DD, DD);

    // 8) LN2 -> bf16
    ln_kernel<<<MM, 256, 0, stream>>>(x1, g2, bt2, x2b, nullptr, nullptr, nullptr, nullptr,
                                      nullptr, nullptr);

    // 9) FFN up + ReLU -> bf16
    dim3 gF1(MM / 32, FFD / 256);
    gemm_wmma<2><<<gF1, 128, 0, stream>>>(x2b, W1T, b1, nullptr, nullptr, hb, MM, FFD, DD);

    // 10) FFN down + residual -> d_out
    gemm_wmma<1><<<gQ, 128, 0, stream>>>(hb, W2T, b2, x1, out, nullptr, MM, DD, FFD);
}
